// MCBFusion_72799695667458
// MI455X (gfx1250) — compile-verified, matrix operand order checked by reading, not run on hardware
//
#include <hip/hip_runtime.h>
#include <hip/hip_bf16.h>
#include <stdint.h>

// ---------------------------------------------------------------------------
// MCB fusion pipeline for MI455X (gfx1250, wave32).
//
// Roofline: conv1 GEMM (1568x512x16000 = 25.7 GFLOP) dominates -> bf16 WMMA
// (v_wmma_f32_16x16x32_bf16) with f32 accumulate. CBP circular convolution is
// done directly from LDS-resident count sketches (64KB/row fits in the 320KB
// WGP LDS) instead of an FFT; sketches built with ds_add_f32 LDS atomics.
// Global->LDS staging uses CDNA5 async loads (global_load_async_to_lds_b128,
// ASYNCcnt) with double buffering in the GEMM.
// ---------------------------------------------------------------------------

typedef __attribute__((ext_vector_type(16))) __bf16 v16bf;
typedef __attribute__((ext_vector_type(8)))  float  v8f;

#define MCB_D     16000
#define IMG_C     2048
#define NPOS      49          // 7x7
#define BS        32
#define P_A       (BS * NPOS) // 1568

// ---------------------------------------------------------------------------
// CDNA5 async global->LDS helpers.
// AS3 flat addresses keep the LDS byte offset in the low 32 bits (aperture in
// [63:32]), so truncation yields the VGPR value global_load_async_to_lds wants.
// ---------------------------------------------------------------------------
__device__ __forceinline__ uint32_t lds_off(const void* p) {
  return (uint32_t)(uintptr_t)p;
}
__device__ __forceinline__ void async_copy16(uint32_t lds_dst, const void* g_src) {
  asm volatile("global_load_async_to_lds_b128 %0, %1, off"
               :: "v"(lds_dst), "v"(g_src) : "memory");
}
__device__ __forceinline__ void async_wait0() {
  asm volatile("s_wait_asynccnt 0" ::: "memory");
}

// ---------------------------------------------------------------------------
// K1: AvgPool2d((3,3), stride=(100,1), pad=(1,1), count_include_pad) on
// txt (32,50,2048) -> txt_p (32,2048). Only one output row (rows 0,1 valid).
// ---------------------------------------------------------------------------
__global__ void avgpool_txt_kernel(const float* __restrict__ txt,
                                   float* __restrict__ txt_p) {
  int b = blockIdx.y;
  int j = blockIdx.x * blockDim.x + threadIdx.x;
  if (j >= IMG_C) return;
  const float* t = txt + (size_t)b * 50 * IMG_C;
  float s = 0.f;
  #pragma unroll
  for (int r = 0; r < 2; ++r)
    for (int c = j - 1; c <= j + 1; ++c)
      if (c >= 0 && c < IMG_C) s += t[r * IMG_C + c];
  txt_p[(size_t)b * IMG_C + j] = s * (1.0f / 9.0f);
}

// ---------------------------------------------------------------------------
// K2: a1[p,i] = s1[i] * txt_p[b, (i*S + wh) % C]   with p = b*S + wh.
// S=49 reproduces the wrap-around repeat/reshape tiling; S=1 gives a1B.
// ---------------------------------------------------------------------------
__global__ void build_a1_kernel(const float* __restrict__ txt_p,
                                const int* __restrict__ s1,
                                float* __restrict__ a1, int S) {
  int p = blockIdx.x;
  int b  = p / S;
  int wh = p % S;
  for (int i = threadIdx.x; i < IMG_C; i += blockDim.x) {
    int src = (i * S + wh) % IMG_C;
    a1[(size_t)p * IMG_C + i] =
        (float)s1[i] * txt_p[(size_t)b * IMG_C + src];
  }
}

// ---------------------------------------------------------------------------
// K3: count sketch per position, built in LDS with ds_add_f32 atomics.
// mode 0: x2 is NCHW img, element (p=(b,wh), j) at [(p/49)*C*49 + j*49 + p%49]
// mode 1: x2 row-major [P, C]
// ---------------------------------------------------------------------------
__global__ void sketch_kernel(const float* __restrict__ x2,
                              const int* __restrict__ h,
                              const int* __restrict__ s,
                              float* __restrict__ skout, int mode) {
  __shared__ __align__(16) float sk[MCB_D];
  int p = blockIdx.x;
  for (int k = threadIdx.x; k < MCB_D; k += blockDim.x) sk[k] = 0.f;
  __syncthreads();
  for (int j = threadIdx.x; j < IMG_C; j += blockDim.x) {
    float v = mode ? x2[(size_t)p * IMG_C + j]
                   : x2[(size_t)(p / NPOS) * IMG_C * NPOS +
                        (size_t)j * NPOS + (p % NPOS)];
    atomicAdd(&sk[h[j]], v * (float)s[j]);   // -> ds_add_f32
  }
  __syncthreads();
  for (int k = threadIdx.x; k < MCB_D; k += blockDim.x)
    skout[(size_t)p * MCB_D + k] = sk[k];
}

// ---------------------------------------------------------------------------
// K4: CBP circular convolution, sketch2 resident in LDS (async-loaded).
// out[p,k] = sum_i a1[p,i] * sk2[p, (k - h1[i]) mod D]
// Consecutive lanes hit consecutive LDS addresses -> conflict-free gathers.
// ---------------------------------------------------------------------------
__global__ __launch_bounds__(256)
void cbp_circconv_kernel(const float* __restrict__ a1,
                         const float* __restrict__ sk2,
                         const int* __restrict__ h1,
                         float* __restrict__ out) {
  __shared__ __align__(16) float s_sk[MCB_D];
  int p = blockIdx.x;
  const float* skrow = sk2 + (size_t)p * MCB_D;
  for (int c = threadIdx.x; c < MCB_D / 4; c += blockDim.x)
    async_copy16(lds_off(&s_sk[c * 4]), (const void*)(skrow + c * 4));
  async_wait0();
  __syncthreads();

  const float* a1row = a1 + (size_t)p * IMG_C;
  for (int k = threadIdx.x; k < MCB_D; k += blockDim.x) {
    float acc = 0.f;
    #pragma unroll 4
    for (int i = 0; i < IMG_C; ++i) {
      int idx = k - h1[i];
      idx += (idx < 0) ? MCB_D : 0;
      acc = fmaf(a1row[i], s_sk[idx], acc);
    }
    out[(size_t)p * MCB_D + k] = acc;
  }
}

// ---------------------------------------------------------------------------
// K5: signed_sqrt + l2norm over D per row, emit bf16 for the WMMA GEMM.
// signed_sqrt(x)^2 == |x|, so the norm is sqrt(sum |x|).
// ---------------------------------------------------------------------------
__global__ void ssqrt_l2norm_bf16_kernel(const float* __restrict__ in,
                                         __bf16* __restrict__ out) {
  __shared__ float warpsum[8];
  int p = blockIdx.x;
  const float* row = in + (size_t)p * MCB_D;
  float ss = 0.f;
  for (int k = threadIdx.x; k < MCB_D; k += blockDim.x) ss += fabsf(row[k]);
  for (int o = 16; o > 0; o >>= 1) ss += __shfl_down(ss, o, 32);
  if ((threadIdx.x & 31) == 0) warpsum[threadIdx.x >> 5] = ss;
  __syncthreads();
  if (threadIdx.x == 0) {
    float t = 0.f;
    for (int w = 0; w < 8; ++w) t += warpsum[w];
    warpsum[0] = 1.f / fmaxf(sqrtf(t), 1e-12f);
  }
  __syncthreads();
  float inv = warpsum[0];
  for (int k = threadIdx.x; k < MCB_D; k += blockDim.x) {
    float x = row[k];
    float y = copysignf(sqrtf(fabsf(x)), x) * inv;
    out[(size_t)p * MCB_D + k] = (__bf16)y;
  }
}

// ---------------------------------------------------------------------------
// K6: f32 -> bf16 weight conversion (grid-stride).
// ---------------------------------------------------------------------------
__global__ void f32_to_bf16_kernel(const float* __restrict__ in,
                                   __bf16* __restrict__ out, long n) {
  long i = (long)blockIdx.x * blockDim.x + threadIdx.x;
  long stride = (long)gridDim.x * blockDim.x;
  for (; i < n; i += stride) out[i] = (__bf16)in[i];
}

// ---------------------------------------------------------------------------
// K7: WMMA GEMM  C[M,N] = A[M,K](bf16) x B[N,K]^T(bf16) + bias, opt relu.
// 256 threads = 8 waves; block tile 128x64; wave tile 32x32 (2x2 WMMA 16x16);
// BK=32 (one v_wmma_f32_16x16x32_bf16 per tile per k-step). Double-buffered
// LDS staging via global_load_async_to_lds_b128 + s_wait_asynccnt.
// Out-of-range rows are clamped (their outputs are never stored).
// ---------------------------------------------------------------------------
#define BM 128
#define BN 64
#define BK 32
#define LDA 48   // padded row length (bf16 elems) = 96B, 16B aligned

__global__ __launch_bounds__(256)
void gemm_bf16_wmma_kernel(const __bf16* __restrict__ A,
                           const __bf16* __restrict__ B,
                           const float* __restrict__ bias,
                           float* __restrict__ C,
                           int M, int N, int K, int relu) {
  __shared__ __align__(16) __bf16 As[2][BM * LDA];
  __shared__ __align__(16) __bf16 Bs[2][BN * LDA];

  const int tid    = threadIdx.x;
  const int lane   = tid & 31;
  const int wave   = tid >> 5;
  const int half   = lane >> 4;      // K-half for A/B frags, M+8 for C
  const int r      = lane & 15;
  const int wmBase = (wave & 3) * 32;
  const int wnBase = (wave >> 2) * 32;
  const int blockM = blockIdx.y * BM;
  const int blockN = blockIdx.x * BN;

  uint32_t ldsA[2] = { lds_off(&As[0][0]), lds_off(&As[1][0]) };
  uint32_t ldsB[2] = { lds_off(&Bs[0][0]), lds_off(&Bs[1][0]) };

  auto issue = [&](int buf, int k0) {
    #pragma unroll
    for (int rep = 0; rep < 2; ++rep) {          // A: 128 rows x 4 chunks
      int c   = tid + rep * 256;
      int row = c >> 2, q = c & 3;
      int gm  = blockM + row; if (gm > M - 1) gm = M - 1;
      const __bf16* gp = A + (size_t)gm * K + (k0 + q * 8);
      async_copy16(ldsA[buf] + (uint32_t)(row * (LDA * 2) + q * 16), gp);
    }
    {                                            // B: 64 rows x 4 chunks
      int row = tid >> 2, q = tid & 3;
      int gn  = blockN + row; if (gn > N - 1) gn = N - 1;
      const __bf16* gp = B + (size_t)gn * K + (k0 + q * 8);
      async_copy16(ldsB[buf] + (uint32_t)(row * (LDA * 2) + q * 16), gp);
    }
  };

  v8f acc[2][2];
  #pragma unroll
  for (int i = 0; i < 2; ++i)
    #pragma unroll
    for (int j = 0; j < 2; ++j)
      #pragma unroll
      for (int e = 0; e < 8; ++e) acc[i][j][e] = 0.f;

  const int KT = K / BK;
  issue(0, 0);

  for (int kt = 0; kt < KT; ++kt) {
    int cur = kt & 1;
    async_wait0();
    __syncthreads();
    if (kt + 1 < KT) issue(cur ^ 1, (kt + 1) * BK);

    v16bf af[2], bfr[2];
    #pragma unroll
    for (int i = 0; i < 2; ++i) {
      const __bf16* arow = &As[cur][(wmBase + i * 16 + r) * LDA];
      uint4* av = reinterpret_cast<uint4*>(&af[i]);
      av[0] = *reinterpret_cast<const uint4*>(arow + half * 8);        // K 0..7 / 8..15
      av[1] = *reinterpret_cast<const uint4*>(arow + 16 + half * 8);   // K 16..23 / 24..31
    }
    #pragma unroll
    for (int j = 0; j < 2; ++j) {
      const __bf16* brow = &Bs[cur][(wnBase + j * 16 + r) * LDA];
      uint4* bv = reinterpret_cast<uint4*>(&bfr[j]);
      bv[0] = *reinterpret_cast<const uint4*>(brow + half * 8);
      bv[1] = *reinterpret_cast<const uint4*>(brow + 16 + half * 8);
    }
    #pragma unroll
    for (int i = 0; i < 2; ++i)
      #pragma unroll
      for (int j = 0; j < 2; ++j)
        acc[i][j] = __builtin_amdgcn_wmma_f32_16x16x32_bf16(
            false, af[i], false, bfr[j], (short)0, acc[i][j], false, false);
    __syncthreads();
  }

  // C/D layout: VGPR e -> M = e (+8 for lanes 16..31), N = lane&15.
  #pragma unroll
  for (int i = 0; i < 2; ++i)
    #pragma unroll
    for (int j = 0; j < 2; ++j) {
      int n = blockN + wnBase + j * 16 + r;
      #pragma unroll
      for (int e = 0; e < 8; ++e) {
        int m = blockM + wmBase + i * 16 + half * 8 + e;
        if (m < M && n < N) {
          float v = acc[i][j][e] + bias[n];
          if (relu) v = fmaxf(v, 0.f);
          C[(size_t)m * N + n] = v;
        }
      }
    }
}

// ---------------------------------------------------------------------------
// K8: conv2 (512->1) + softmax over the 49 positions of each batch.
// ---------------------------------------------------------------------------
__global__ void attn_kernel(const float* __restrict__ C1,
                            const float* __restrict__ w2,
                            const float* __restrict__ b2,
                            float* __restrict__ att) {
  __shared__ float lg[NPOS];
  int b = blockIdx.x;
  int p = threadIdx.x;   // blockDim 64
  if (p < NPOS) {
    const float* row = C1 + ((size_t)b * NPOS + p) * 512;
    float s = b2[0];
    for (int o = 0; o < 512; ++o) s = fmaf(row[o], w2[o], s);
    lg[p] = s;
  }
  __syncthreads();
  if (threadIdx.x == 0) {
    float mx = lg[0];
    for (int q = 1; q < NPOS; ++q) mx = fmaxf(mx, lg[q]);
    float sum = 0.f;
    for (int q = 0; q < NPOS; ++q) { lg[q] = expf(lg[q] - mx); sum += lg[q]; }
    float inv = 1.f / sum;
    for (int q = 0; q < NPOS; ++q) att[(size_t)b * NPOS + q] = lg[q] * inv;
  }
}

// ---------------------------------------------------------------------------
// K9: torch-faithful .view(bs,49,-1) weighted pool:
// res[b,c] = sum_p img_flat[b][p*2048 + c] * att[b,p]
// ---------------------------------------------------------------------------
__global__ void wpool_kernel(const float* __restrict__ img,
                             const float* __restrict__ att,
                             float* __restrict__ res) {
  int b = blockIdx.y;
  int c = blockIdx.x * blockDim.x + threadIdx.x;
  if (c >= IMG_C) return;
  const float* base = img + (size_t)b * IMG_C * NPOS;
  float s = 0.f;
  #pragma unroll
  for (int p = 0; p < NPOS; ++p)
    s = fmaf(base[(size_t)p * IMG_C + c], att[(size_t)b * NPOS + p], s);
  res[(size_t)b * IMG_C + c] = s;
}

// ---------------------------------------------------------------------------
// Host side: workspace bump allocator + launch sequence.
// ---------------------------------------------------------------------------
static inline char* bump(char*& p, size_t bytes) {
  char* r = p;
  p += (bytes + 255) & ~(size_t)255;
  return r;
}

extern "C" void kernel_launch(void* const* d_in, const int* in_sizes, int n_in,
                              void* d_out, int out_size, void* d_ws, size_t ws_size,
                              hipStream_t stream) {
  (void)in_sizes; (void)n_in; (void)out_size; (void)ws_size;

  const float* txt     = (const float*)d_in[0];
  const float* img     = (const float*)d_in[1];
  const float* conv1_w = (const float*)d_in[2];
  const float* conv1_b = (const float*)d_in[3];
  const float* conv2_w = (const float*)d_in[4];
  const float* conv2_b = (const float*)d_in[5];
  const float* lin2_w  = (const float*)d_in[6];
  const float* lin2_b  = (const float*)d_in[7];
  const int* h1a = (const int*)d_in[8];  const int* s1a = (const int*)d_in[9];
  const int* h2a = (const int*)d_in[10]; const int* s2a = (const int*)d_in[11];
  const int* h1b = (const int*)d_in[12]; const int* s1b = (const int*)d_in[13];
  const int* h2b = (const int*)d_in[14]; const int* s2b = (const int*)d_in[15];
  float* out = (float*)d_out;

  char* ws = (char*)d_ws;
  float*  txt_p = (float*)bump(ws, (size_t)BS * IMG_C * 4);
  float*  a1A   = (float*)bump(ws, (size_t)P_A * IMG_C * 4);
  char*   shareA = bump(ws, (size_t)P_A * MCB_D * 4);  // sk2A, later lin2_w bf16
  float*  sk2A  = (float*)shareA;
  __bf16* W3bf  = (__bf16*)shareA;                     // 96MB <= 100MB, disjoint in time
  float*  cbpA  = (float*)bump(ws, (size_t)P_A * MCB_D * 4);
  __bf16* actA  = (__bf16*)bump(ws, (size_t)P_A * MCB_D * 2);
  __bf16* W1bf  = (__bf16*)bump(ws, (size_t)512 * MCB_D * 2);
  float*  C1    = (float*)bump(ws, (size_t)P_A * 512 * 4);
  float*  att   = (float*)bump(ws, (size_t)BS * NPOS * 4);
  float*  res   = (float*)bump(ws, (size_t)BS * IMG_C * 4);
  float*  a1B   = (float*)bump(ws, (size_t)BS * IMG_C * 4);
  float*  sk2B  = (float*)bump(ws, (size_t)BS * MCB_D * 4);
  float*  cbpB  = (float*)bump(ws, (size_t)BS * MCB_D * 4);
  __bf16* finbf = (__bf16*)bump(ws, (size_t)BS * MCB_D * 2);

  // K1: avgpool -> txt_p (32,2048)
  avgpool_txt_kernel<<<dim3(8, BS), 256, 0, stream>>>(txt, txt_p);
  // K2: tiled signed txt features a1A (1568,2048)
  build_a1_kernel<<<P_A, 256, 0, stream>>>(txt_p, s1a, a1A, NPOS);
  // K3: img count sketches sk2A (1568,16000)
  sketch_kernel<<<P_A, 256, 0, stream>>>(img, h2a, s2a, sk2A, 0);
  // K4: CBP-A circular convolution (LDS-resident sketch, async fill)
  cbp_circconv_kernel<<<P_A, 256, 0, stream>>>(a1A, sk2A, h1a, cbpA);
  // K5: signed_sqrt + l2norm -> bf16 activations
  ssqrt_l2norm_bf16_kernel<<<P_A, 256, 0, stream>>>(cbpA, actA);
  // K6: conv1 weights -> bf16
  f32_to_bf16_kernel<<<4096, 256, 0, stream>>>(conv1_w, W1bf, (long)512 * MCB_D);
  // K7: conv1 GEMM (WMMA bf16) + relu : (1568,16000)x(512,16000)^T
  gemm_bf16_wmma_kernel<<<dim3((512 + BN - 1) / BN, (P_A + BM - 1) / BM),
                          256, 0, stream>>>(actA, W1bf, conv1_b, C1,
                                            P_A, 512, MCB_D, 1);
  // K8: conv2 + softmax attention (32,49)
  attn_kernel<<<BS, 64, 0, stream>>>(C1, conv2_w, conv2_b, att);
  // K9: attention-weighted pooling -> res (32,2048)
  wpool_kernel<<<dim3(8, BS), 256, 0, stream>>>(img, att, res);
  // K10: a1B = s1b * txt_p (32,2048)
  build_a1_kernel<<<BS, 256, 0, stream>>>(txt_p, s1b, a1B, 1);
  // K11: res count sketches sk2B (32,16000)
  sketch_kernel<<<BS, 256, 0, stream>>>(res, h2b, s2b, sk2B, 1);
  // K12: CBP-B circular convolution (32,16000)
  cbp_circconv_kernel<<<BS, 256, 0, stream>>>(a1B, sk2B, h1b, cbpB);
  // K13: signed_sqrt + l2norm -> bf16
  ssqrt_l2norm_bf16_kernel<<<BS, 256, 0, stream>>>(cbpB, finbf);
  // K14: lin2 weights -> bf16 (reuses sk2A region, dead after K4)
  f32_to_bf16_kernel<<<65536, 256, 0, stream>>>(lin2_w, W3bf, (long)3000 * MCB_D);
  // K15: lin2 GEMM (WMMA bf16): (32,16000)x(3000,16000)^T -> d_out (32,3000)
  gemm_bf16_wmma_kernel<<<dim3((3000 + BN - 1) / BN, (BS + BM - 1) / BM),
                          256, 0, stream>>>(finbf, W3bf, lin2_b, out,
                                            BS, 3000, MCB_D, 0);
}
